// GCN_33672543600970
// MI455X (gfx1250) — matrix operand order repaired and checked by reading, compiled, not run on hardware
//
#include <hip/hip_runtime.h>
#include <hip/hip_bf16.h>

// ---------------------------------------------------------------------------
// GCN (3x GCNConv + linear + softmax) for MI455X (gfx1250, wave32, WMMA).
//   * bf16 WMMA (v_wmma_f32_16x16x32_bf16) for all GEMMs, f32 accumulate.
//   * A-tile staged to LDS once per block via gfx1250 async-to-LDS DMA
//     (global_load_async_to_lds_b128 / s_wait_asynccnt), fragments read with
//     ds_load_b128 -> 8x less global A traffic than per-wave loads.
//   * Graph data (h: 25.6MB, edges: 6.4MB) fits in 192MB L2 -> scatter runs
//     at L2 speed via coalesced global_atomic_add_f32 + global_prefetch_b8.
// ---------------------------------------------------------------------------

typedef __attribute__((ext_vector_type(16))) __bf16 v16bf;
typedef __attribute__((ext_vector_type(8)))  float  v8f;

union Frag16 {
    v16bf v;
    uint4 q[2];
};

#define NFEAT  128

// ---------------------------------------------------------------- norm setup
__global__ void k_deg_init(float* __restrict__ deg, int n) {
    int i = blockIdx.x * blockDim.x + threadIdx.x;
    if (i < n) deg[i] = 1.0f;  // self-loop contributes +1
}

__global__ void k_deg_count(const int* __restrict__ dst, float* __restrict__ deg, int e) {
    int i = blockIdx.x * blockDim.x + threadIdx.x;
    if (i < e) atomicAdd(&deg[dst[i]], 1.0f);
}

__global__ void k_dinv(float* __restrict__ deg_to_dinv, int n) {
    int i = blockIdx.x * blockDim.x + threadIdx.x;
    if (i < n) deg_to_dinv[i] = rsqrtf(deg_to_dinv[i]);
}

// ------------------------------------------------- weight convert + transpose
// W: [K x OUTW] row-major f32  ->  Wt: [OUTW x K] (K contiguous) bf16
__global__ void k_wconv(const float* __restrict__ W, __bf16* __restrict__ Wt,
                        int K, int OUTW) {
    int i = blockIdx.x * blockDim.x + threadIdx.x;
    if (i >= K * OUTW) return;
    int k = i / OUTW;
    int c = i - k * OUTW;
    Wt[(size_t)c * K + k] = (__bf16)W[i];
}

// --------------------------------------------------- activation f32 -> bf16
__global__ void k_hconv(const float* __restrict__ src, __bf16* __restrict__ dst,
                        int n, int relu) {
    int stride = gridDim.x * blockDim.x;
    for (int i = blockIdx.x * blockDim.x + threadIdx.x; i < n; i += stride) {
        float v = src[i];
        if (relu) v = fmaxf(v, 0.0f);
        dst[i] = (__bf16)v;
    }
}

// ------------------------------------------------------------- WMMA GEMM
// C[M x OUT] = A[M x 128](bf16, row-major) * Bt[OUT x 128](bf16, K contiguous)
// One block = 16 rows; wave w computes N-tile w (16 cols). OUT/16 waves/block.
// A tile (16x128 bf16 = 4KB) staged into LDS with async-to-LDS DMA.
template <int OUT>
__global__ void k_gemm(const __bf16* __restrict__ A, const __bf16* __restrict__ Bt,
                       const float* __restrict__ bias, float* __restrict__ C) {
    __shared__ __bf16 sA[16 * 128];                 // 4096 B

    const int tid   = threadIdx.x;                  // blockDim = OUT*2
    const int lane  = tid & 31;
    const int wave  = tid >> 5;                     // N-tile index
    const int r     = lane & 15;
    const int hi    = lane >> 4;                    // 0: lanes 0-15, 1: 16-31
    const int mbase = blockIdx.x * 16;

    // ---- async stage: 256 chunks of 16B; each thread NLOADS chunks --------
    constexpr int NTHREADS = OUT * 2;
    constexpr int NLOADS   = 256 / NTHREADS;        // 128->1, 64->2
    const uint32_t lds0 = (uint32_t)(uintptr_t)(&sA[0]);
    const __bf16* gtile = A + (size_t)mbase * 128;
#pragma unroll
    for (int j = 0; j < NLOADS; ++j) {
        const int chunk = tid + j * NTHREADS;       // 16B chunk index
        const uint32_t ldst = lds0 + chunk * 16;
        const __bf16* gsrc = gtile + chunk * 8;
        asm volatile("global_load_async_to_lds_b128 %0, %1, off"
                     :: "v"(ldst), "v"(gsrc)
                     : "memory");
    }
    asm volatile("s_wait_asynccnt 0x0" ::: "memory");
    __syncthreads();

    // ---- WMMA over K = 4 x 32 --------------------------------------------
    const __bf16* arow = &sA[r * 128];
    const __bf16* bcol = Bt + (size_t)(wave * 16 + r) * 128;

    v8f acc = {};
#pragma unroll
    for (int ks = 0; ks < 4; ++ks) {
        const int kb = ks * 32;
        Frag16 a, b;
        // A 16x32 fragment from LDS (ds_load_b128): lanes 0-15 hold
        // K {0..7,16..23}, lanes 16-31 hold K {8..15,24..31} of row r.
        a.q[0] = *(const uint4*)(arow + kb + hi * 8);
        a.q[1] = *(const uint4*)(arow + kb + hi * 8 + 16);
        // B 32x16 fragment: lane holds column (wave*16+r), same K split as A.
        b.q[0] = *(const uint4*)(bcol + kb + hi * 8);
        b.q[1] = *(const uint4*)(bcol + kb + hi * 8 + 16);
        acc = __builtin_amdgcn_wmma_f32_16x16x32_bf16(
            false, a.v, false, b.v, (short)0, acc, false, false);
    }

    const int col = wave * 16 + r;
    const float bv = bias ? bias[col] : 0.0f;
    float* cp = C + (size_t)mbase * OUT + col;
#pragma unroll
    for (int v = 0; v < 8; ++v) {
        const int m = v + hi * 8;                   // C layout: VGPR v -> M=v(+8)
        cp[(size_t)m * OUT] = acc[v] + bv;
    }
}

// ------------------------------------- h_next = ht * dinv^2 + b  (self loop)
__global__ void k_init_next(const float* __restrict__ ht, const float* __restrict__ dinv,
                            const float* __restrict__ b, float* __restrict__ hn, int n) {
    int i = blockIdx.x * blockDim.x + threadIdx.x;
    if (i >= n) return;
    int row = i >> 7;
    int col = i & 127;
    float dv = dinv[row];
    hn[i] = ht[i] * dv * dv + b[col];
}

// ----------------------------------------------------- edge scatter-add (L2)
// 256 threads: 2 edges per block, 128 coalesced f32 atomics per edge.
__global__ void k_edge(const int* __restrict__ src, const int* __restrict__ dst,
                       const float* __restrict__ dinv, const float* __restrict__ ht,
                       float* __restrict__ hn, int e) {
    const int f    = threadIdx.x & 127;
    const int esub = threadIdx.x >> 7;
    const int i    = blockIdx.x * 2 + esub;
    if (i >= e) return;
    const int s = src[i];
    const int d = dst[i];
    // Prefetch next edge's source row into cache (global_prefetch_b8).
    if (i + 2 < e) {
        __builtin_prefetch(&ht[(size_t)src[i + 2] * 128 + f], 0, 1);
    }
    const float w = dinv[s] * dinv[d];
    atomicAdd(&hn[(size_t)d * 128 + f], ht[(size_t)s * 128 + f] * w);
}

// ------------------------------------------------- row softmax over 64 cols
// One wave per row: lane holds cols {lane, lane+32}; shfl_xor reductions.
__global__ void k_softmax(float* __restrict__ out) {
    const int row  = blockIdx.x;
    const int lane = threadIdx.x;
    float* p = out + (size_t)row * 64;
    float v0 = p[lane];
    float v1 = p[lane + 32];
    float m = fmaxf(v0, v1);
#pragma unroll
    for (int off = 16; off > 0; off >>= 1)
        m = fmaxf(m, __shfl_xor(m, off, 32));
    float e0 = __expf(v0 - m);
    float e1 = __expf(v1 - m);
    float s = e0 + e1;
#pragma unroll
    for (int off = 16; off > 0; off >>= 1)
        s += __shfl_xor(s, off, 32);
    float inv = 1.0f / s;
    p[lane]      = e0 * inv;
    p[lane + 32] = e1 * inv;
}

// ---------------------------------------------------------------------------
extern "C" void kernel_launch(void* const* d_in, const int* in_sizes, int n_in,
                              void* d_out, int out_size, void* d_ws, size_t ws_size,
                              hipStream_t stream) {
    const float* x  = (const float*)d_in[0];
    const int*   ei = (const int*)d_in[1];      // [2, E] int32
    const float* W0 = (const float*)d_in[2];
    const float* b0 = (const float*)d_in[3];
    const float* W1 = (const float*)d_in[4];
    const float* b1 = (const float*)d_in[5];
    const float* W2 = (const float*)d_in[6];
    const float* b2 = (const float*)d_in[7];
    const float* Wl = (const float*)d_in[8];
    const float* bl = (const float*)d_in[9];
    float* out = (float*)d_out;

    const int N = in_sizes[0] / NFEAT;          // 50000
    const int E = in_sizes[1] / 2;              // 800000
    const int* src = ei;
    const int* dst = ei + E;

    // ---- workspace carve-up (~64.4 MB) -----------------------------------
    char* ws = (char*)d_ws;
    float*  ht   = (float*)ws;                         // N*128 f32
    float*  hn   = ht + (size_t)N * 128;               // N*128 f32
    __bf16* hb   = (__bf16*)(hn + (size_t)N * 128);    // N*128 bf16
    float*  dinv = (float*)(hb + (size_t)N * 128);     // N f32
    __bf16* wt0  = (__bf16*)(dinv + N);                // 128*128 bf16 each
    __bf16* wt1  = wt0 + 128 * 128;
    __bf16* wt2  = wt1 + 128 * 128;
    __bf16* wtl  = wt2 + 128 * 128;                    // 64*128 bf16

    const int TB = 256;
    const int nM = N / 16;                             // 3125 M-tiles

    // ---- gcn_norm: deg -> dinv ------------------------------------------
    k_deg_init<<<(N + TB - 1) / TB, TB, 0, stream>>>(dinv, N);
    k_deg_count<<<(E + TB - 1) / TB, TB, 0, stream>>>(dst, dinv, E);
    k_dinv<<<(N + TB - 1) / TB, TB, 0, stream>>>(dinv, N);

    // ---- weights -> transposed bf16 -------------------------------------
    k_wconv<<<(128 * 128 + TB - 1) / TB, TB, 0, stream>>>(W0, wt0, 128, 128);
    k_wconv<<<(128 * 128 + TB - 1) / TB, TB, 0, stream>>>(W1, wt1, 128, 128);
    k_wconv<<<(128 * 128 + TB - 1) / TB, TB, 0, stream>>>(W2, wt2, 128, 128);
    k_wconv<<<(128 * 64  + TB - 1) / TB, TB, 0, stream>>>(Wl, wtl, 128, 64);

    // ---- 3 GCN layers ----------------------------------------------------
    const __bf16* wts[3]    = {wt0, wt1, wt2};
    const float*  biases[3] = {b0, b1, b2};
    const int nElem = N * 128;
    for (int layer = 0; layer < 3; ++layer) {
        const float* hin = (layer == 0) ? x : hn;
        k_hconv<<<2048, TB, 0, stream>>>(hin, hb, nElem, layer > 0 ? 1 : 0);
        k_gemm<128><<<nM, 256, 0, stream>>>(hb, wts[layer], nullptr, ht);
        k_init_next<<<(nElem + TB - 1) / TB, TB, 0, stream>>>(ht, dinv, biases[layer], hn, nElem);
        k_edge<<<(E + 1) / 2, 256, 0, stream>>>(src, dst, dinv, ht, hn, E);
    }

    // ---- classifier + softmax -------------------------------------------
    k_hconv<<<2048, TB, 0, stream>>>(hn, hb, nElem, 1);
    k_gemm<64><<<nM, 128, 0, stream>>>(hb, wtl, bl, out);
    k_softmax<<<N, 32, 0, stream>>>(out);
}
